// Mapper_12232066859066
// MI455X (gfx1250) — matrix operand order repaired and checked by reading, not compiled, not run on hardware
//
#include <hip/hip_runtime.h>
#include <hip/hip_bf16.h>
#include <math.h>

#define B_ 8
#define L_ 4096
#define F_ 512
#define H_ 512
#define P_ 256
#define DEPTH_ 6
#define M_ (B_*L_)      // 32768 rows
#define NCH 32          // scan chunks
#define CLEN (L_/NCH)   // 128 steps per chunk

typedef __attribute__((ext_vector_type(16))) __bf16 v16bf;
typedef __attribute__((ext_vector_type(8)))  float  v8f;

// ---------- helpers ----------
__device__ __forceinline__ unsigned short f2b(float x) {
    unsigned u = __float_as_uint(x);
    unsigned r = u + 0x7FFFu + ((u >> 16) & 1u);   // round-to-nearest-even bf16
    return (unsigned short)(r >> 16);
}
__device__ __forceinline__ float gelu_tanh(float x) {
    float x3 = x * x * x;
    return 0.5f * x * (1.0f + tanhf(0.7978845608028654f * (x + 0.044715f * x3)));
}

struct C2 { float r, i; };
__device__ __forceinline__ C2 cmul(C2 a, C2 b) {
    C2 o; o.r = a.r * b.r - a.i * b.i; o.i = a.r * b.i + a.i * b.r; return o;
}

// ---------- one-time prep kernels ----------
// transpose + convert: src [D][R][C] f32 row-major -> dst [D][C][R] bf16
__global__ void cvt_bf16_t_batched(const float* __restrict__ src, unsigned short* __restrict__ dst,
                                   int R, int C, int D) {
    long i = (long)blockIdx.x * blockDim.x + threadIdx.x;
    if (i >= (long)D * R * C) return;
    int c = (int)(i % C);
    long t = i / C;
    int r = (int)(t % R);
    int d = (int)(t / R);
    dst[((long)d * C + c) * R + r] = f2b(src[i]);
}

// Build per-layer discretized SSM weights, stored TRANSPOSED ([N][K]) for the GEMM:
//  Wbut[l][2p][h],[2p+1][h] = Re/Im( ((lam_bar-1)/lam) * (B_re + i B_im)[p][h] )
//     -> u @ Wbut^T = Bu interleaved (N=2P, K=H)
//  Cwt[l][h][2p] = C_re[h][p] ; Cwt[l][h][2p+1] = -C_im[h][p]
//     -> x @ Cwt^T = Re(x C)   (N=H, K=2P)
__global__ void build_weights(const float* __restrict__ lam_re, const float* __restrict__ lam_im,
                              const float* __restrict__ log_dt,
                              const float* __restrict__ B_re, const float* __restrict__ B_im,
                              const float* __restrict__ C_re, const float* __restrict__ C_im,
                              unsigned short* __restrict__ Wbut, unsigned short* __restrict__ Cwt,
                              float* __restrict__ lamBar) {
    long idx = (long)blockIdx.x * blockDim.x + threadIdx.x;
    if (idx >= (long)DEPTH_ * P_ * H_) return;
    int h = (int)(idx % H_);
    long t = idx / H_;
    int p = (int)(t % P_);
    int l = (int)(t / P_);

    float lre = fminf(lam_re[l * P_ + p], -1e-4f);
    float lim = lam_im[l * P_ + p];
    float dt  = expf(log_dt[l * P_ + p]);
    float mag = expf(lre * dt);
    float lbr = mag * cosf(lim * dt);
    float lbi = mag * sinf(lim * dt);
    // f = (lam_bar - 1) / lam
    float nr = lbr - 1.0f, ni = lbi;
    float den = lre * lre + lim * lim;
    float fr = (nr * lre + ni * lim) / den;
    float fi = (ni * lre - nr * lim) / den;

    float br = B_re[((long)l * P_ + p) * H_ + h];
    float bi = B_im[((long)l * P_ + p) * H_ + h];
    float wr = fr * br - fi * bi;
    float wi = fr * bi + fi * br;
    Wbut[((long)l * 2 * P_ + 2 * p    ) * H_ + h] = f2b(wr);
    Wbut[((long)l * 2 * P_ + 2 * p + 1) * H_ + h] = f2b(wi);

    float cr = C_re[((long)l * H_ + h) * P_ + p];
    float ci = C_im[((long)l * H_ + h) * P_ + p];
    Cwt[((long)l * H_ + h) * (2 * P_) + 2 * p    ] = f2b(cr);
    Cwt[((long)l * H_ + h) * (2 * P_) + 2 * p + 1] = f2b(-ci);

    if (h == 0) { lamBar[(l * P_ + p) * 2] = lbr; lamBar[(l * P_ + p) * 2 + 1] = lbi; }
}

// FiLM scale/shift tables: [DEPTH][B][H], fourier activation (even=cos, odd=sin)
__global__ void modfilm(const float* __restrict__ diffr, const float* __restrict__ favr,
                        const float* __restrict__ shift_W, const float* __restrict__ shift_b,
                        const float* __restrict__ scale_W, const float* __restrict__ scale_b,
                        float* __restrict__ shiftArr, float* __restrict__ scaleArr) {
    long idx = (long)blockIdx.x * blockDim.x + threadIdx.x;
    if (idx >= (long)DEPTH_ * B_ * H_) return;
    int h = (int)(idx % H_);
    long t = idx / H_;
    int b = (int)(t % B_);
    int l = (int)(t / B_);
    float r0 = diffr[b], r1 = favr[b];
    float sv = r0 * shift_W[((long)l * 2 + 0) * H_ + h] + r1 * shift_W[((long)l * 2 + 1) * H_ + h]
             + shift_b[(long)l * H_ + h];
    float cv = r0 * scale_W[((long)l * 2 + 0) * H_ + h] + r1 * scale_W[((long)l * 2 + 1) * H_ + h]
             + scale_b[(long)l * H_ + h];
    shiftArr[((long)l * B_ + b) * H_ + h] = (h & 1) ? sinf(sv) : cosf(sv);
    scaleArr[((long)l * B_ + b) * H_ + h] = (h & 1) ? sinf(cv) : cosf(cv);
}

__global__ void mk_reset(const int* __restrict__ seq, int* __restrict__ rst) {
    long i = (long)blockIdx.x * blockDim.x + threadIdx.x;
    if (i >= (long)B_ * L_) return;
    int t = (int)(i % L_);
    int cur = seq[i];
    int prev = (t == 0) ? 0 : seq[i - 1];
    rst[i] = (cur != prev) ? 1 : 0;
}

// ---------- WMMA GEMM: C[M,N] = epi(A[M,K] @ Bt[N,K]^T) ----------
// Block tile 64(M) x 128(N), BK=64. 8 waves arranged 2(M) x 4(N); each wave owns a
// 32x32 output tile = 2x2 16x16 accumulators (4 WMMAs per 32-K slab, 8 ds_load_b128).
// A tile converts f32->bf16 through VGPRs; B tile (pre-transposed bf16 weights) is
// staged with GLOBAL_LOAD_ASYNC_TO_LDS_B128 (ASYNCcnt-tracked, no VGPR round trip).
// EPI 0: acc + bias          EPI 1: gelu(acc + bias)
// EPI 2: u + gelu(acc + dvec[col]*u)   (u = aux)
// EPI 3: aux + acc + bias    (residual add)
template <int EPI, bool HB>
__global__ __launch_bounds__(256)
void gemm_wmma(const float* __restrict__ A, const unsigned short* __restrict__ Bt,
               const float* __restrict__ bias, const float* __restrict__ aux,
               const float* __restrict__ dvec, float* __restrict__ C,
               int M, int N, int K) {
    __shared__ __align__(16) unsigned short As[64][72];    // 64 rows x 64 K (+8 pad)
    __shared__ __align__(16) unsigned short Bs[128][72];   // 128 cols x 64 K (+8 pad)

    int tid  = threadIdx.x;
    int lane = tid & 31;
    int wave = tid >> 5;          // 8 waves
    int waveM = wave & 1;         // 2 along M (32 rows each)
    int waveN = wave >> 1;        // 4 along N (32 cols each)
    int half = lane >> 4;
    int lr   = lane & 15;
    long gm0 = (long)blockIdx.y * 64;
    long gn0 = (long)blockIdx.x * 128;

    int kb  = half * 8;           // A frag: lanes0-15 K0-7/16-23, lanes16-31 K8-15/24-31
    int kb2 = half * 8;           // B frag: lanes0-15 K0-7/16-23, lanes16-31 K8-15/24-31

    v8f acc[2][2] = {{{}, {}}, {{}, {}}};

    for (int k0 = 0; k0 < K; k0 += 64) {
        // stage B tile 128x64 bf16 via async global->LDS (B128 per lane)
        const unsigned short* Bb = Bt + gn0 * K + k0;
#pragma unroll
        for (int it = 0; it < 4; ++it) {
            int q = tid + it * 256;            // 1024 octs of 8 bf16 (16B)
            int r = q >> 3, c8 = (q & 7) * 8;
            unsigned ldsoff = (unsigned)(uintptr_t)(&Bs[r][c8]);
            unsigned long long ga = (unsigned long long)(uintptr_t)(Bb + (long)r * K + c8);
            asm volatile("global_load_async_to_lds_b128 %0, %1, off"
                         :: "v"(ldsoff), "v"(ga) : "memory");
        }
        // stage A tile 64x64 f32->bf16: each thread 4x (float4 load -> ds_store_b64)
        const float* Ab = A + gm0 * K + k0;
#pragma unroll
        for (int it = 0; it < 4; ++it) {
            int q = tid + it * 256;            // 1024 quads
            int r = q >> 4, c4 = (q & 15) * 4;
            float4 f = *(const float4*)(Ab + (long)r * K + c4);
            unsigned lo = (unsigned)f2b(f.x) | ((unsigned)f2b(f.y) << 16);
            unsigned hi = (unsigned)f2b(f.z) | ((unsigned)f2b(f.w) << 16);
            *(uint2*)&As[r][c4] = make_uint2(lo, hi);
        }
        asm volatile("s_wait_asynccnt 0" ::: "memory");
        __syncthreads();

#pragma unroll
        for (int ks = 0; ks < 64; ks += 32) {
            union { v16bf v; uint4 q[2]; } fa[2], fb;
#pragma unroll
            for (int m = 0; m < 2; ++m) {
                int arow = 32 * waveM + 16 * m + lr;
                fa[m].q[0] = *(const uint4*)&As[arow][ks + kb];
                fa[m].q[1] = *(const uint4*)&As[arow][ks + 16 + kb];
            }
#pragma unroll
            for (int n = 0; n < 2; ++n) {
                int bcol = 32 * waveN + 16 * n + lr;
                fb.q[0] = *(const uint4*)&Bs[bcol][ks + kb2];
                fb.q[1] = *(const uint4*)&Bs[bcol][ks + 16 + kb2];
                acc[0][n] = __builtin_amdgcn_wmma_f32_16x16x32_bf16(false, fa[0].v, false, fb.v,
                                                                   (short)0, acc[0][n], false, false);
                acc[1][n] = __builtin_amdgcn_wmma_f32_16x16x32_bf16(false, fa[1].v, false, fb.v,
                                                                   (short)0, acc[1][n], false, false);
            }
        }
        __syncthreads();
    }

    // C layout: VGPR v -> row = half*8 + v (within each 16-row tile), col = lane%16
    long colbase = gn0 + 32 * waveN + lr;
    float bv[2] = {0.0f, 0.0f};
    float dv[2] = {0.0f, 0.0f};
#pragma unroll
    for (int n = 0; n < 2; ++n) {
        if (HB) bv[n] = bias[colbase + 16 * n];
        if (EPI == 2) dv[n] = dvec[colbase + 16 * n];
    }
#pragma unroll
    for (int m = 0; m < 2; ++m) {
#pragma unroll
        for (int v = 0; v < 8; ++v) {
            long row = gm0 + 32 * waveM + 16 * m + half * 8 + v;
            float* cr = C + row * (long)N;
            const float* ar = (EPI >= 2) ? (aux + row * (long)N) : nullptr;
#pragma unroll
            for (int n = 0; n < 2; ++n) {
                long col = colbase + 16 * n;
                float a = acc[m][n][v];
                if (EPI == 0) {
                    cr[col] = a + bv[n];
                } else if (EPI == 1) {
                    cr[col] = gelu_tanh(a + bv[n]);
                } else if (EPI == 2) {
                    float u0 = ar[col];
                    cr[col] = u0 + gelu_tanh(a + dv[n] * u0);
                } else {
                    cr[col] = ar[col] + a + bv[n];
                }
            }
        }
    }
}

// ---------- rowwise layernorm over 512 cols, optionally FiLM affine, in-place safe ----------
template <bool AFFINE>
__global__ __launch_bounds__(256)
void rowln(const float* __restrict__ X, float* __restrict__ Y,
           const float* __restrict__ scaleArr, const float* __restrict__ shiftArr) {
    __shared__ float s1[256], s2[256];
    long row = blockIdx.x;
    int tid = threadIdx.x;
    const float* xr = X + row * 512;
    float a0 = xr[tid], a1 = xr[tid + 256];
    s1[tid] = a0 + a1;
    s2[tid] = a0 * a0 + a1 * a1;
    __syncthreads();
    for (int off = 128; off > 0; off >>= 1) {
        if (tid < off) { s1[tid] += s1[tid + off]; s2[tid] += s2[tid + off]; }
        __syncthreads();
    }
    float mean = s1[0] * (1.0f / 512.0f);
    float var  = s2[0] * (1.0f / 512.0f) - mean * mean;
    float inv  = rsqrtf(var + 1e-5f);
    if (AFFINE) {
        long b = row >> 12;  // row / L_
        const float* sc = scaleArr + b * 512;
        const float* sh = shiftArr + b * 512;
        Y[row * 512 + tid      ] = (a0 - mean) * inv * sc[tid      ] + sh[tid      ];
        Y[row * 512 + tid + 256] = (a1 - mean) * inv * sc[tid + 256] + sh[tid + 256];
    } else {
        Y[row * 512 + tid      ] = (a0 - mean) * inv;
        Y[row * 512 + tid + 256] = (a1 - mean) * inv;
    }
}

// ---------- chunked complex scan: x[t] = a[t]*x[t-1] + Bu[t], a = reset?0:lam_bar ----------
__global__ __launch_bounds__(256)
void scan_phase1(const float* __restrict__ Bu, const int* __restrict__ rst,
                 const float* __restrict__ lamBarL,
                 float* __restrict__ carryA, float* __restrict__ carryX) {
    int p = threadIdx.x;                       // 256 = P_
    int b = blockIdx.x / NCH, ch = blockIdx.x % NCH;
    C2 lb; lb.r = lamBarL[2 * p]; lb.i = lamBarL[2 * p + 1];
    C2 Aacc; Aacc.r = 1.0f; Aacc.i = 0.0f;
    C2 x; x.r = 0.0f; x.i = 0.0f;
    int t0 = ch * CLEN;
    const float2* bp = (const float2*)Bu + ((long)b * L_ + t0) * P_ + p;
    const int* rp = rst + (long)b * L_ + t0;
    for (int t = 0; t < CLEN; ++t) {
        C2 a; if (rp[t]) { a.r = 0.0f; a.i = 0.0f; } else a = lb;
        float2 bu = bp[(long)t * P_];
        x = cmul(a, x); x.r += bu.x; x.i += bu.y;
        Aacc = cmul(a, Aacc);
    }
    long ci = ((long)(b * NCH + ch)) * P_ + p;
    ((float2*)carryA)[ci] = make_float2(Aacc.r, Aacc.i);
    ((float2*)carryX)[ci] = make_float2(x.r, x.i);
}

__global__ void scan_phase2(float* __restrict__ carryA, const float* __restrict__ carryX) {
    long idx = (long)blockIdx.x * blockDim.x + threadIdx.x;
    if (idx >= (long)B_ * P_) return;
    int p = (int)(idx % P_);
    int b = (int)(idx / P_);
    C2 c; c.r = 0.0f; c.i = 0.0f;
    for (int ch = 0; ch < NCH; ++ch) {
        long i = ((long)(b * NCH + ch)) * P_ + p;
        float2 a2 = ((const float2*)carryA)[i];
        float2 x2 = ((const float2*)carryX)[i];
        ((float2*)carryA)[i] = make_float2(c.r, c.i);   // carry entering this chunk
        C2 a; a.r = a2.x; a.i = a2.y;
        C2 t = cmul(a, c);
        c.r = t.r + x2.x; c.i = t.i + x2.y;
    }
}

__global__ __launch_bounds__(256)
void scan_phase3(float* __restrict__ Bu, const int* __restrict__ rst,
                 const float* __restrict__ lamBarL, const float* __restrict__ carryA) {
    int p = threadIdx.x;
    int b = blockIdx.x / NCH, ch = blockIdx.x % NCH;
    C2 lb; lb.r = lamBarL[2 * p]; lb.i = lamBarL[2 * p + 1];
    long ci = ((long)(b * NCH + ch)) * P_ + p;
    float2 c2 = ((const float2*)carryA)[ci];
    C2 x; x.r = c2.x; x.i = c2.y;
    int t0 = ch * CLEN;
    float2* bp = (float2*)Bu + ((long)b * L_ + t0) * P_ + p;
    const int* rp = rst + (long)b * L_ + t0;
    for (int t = 0; t < CLEN; ++t) {
        C2 a; if (rp[t]) { a.r = 0.0f; a.i = 0.0f; } else a = lb;
        float2 bu = bp[(long)t * P_];
        x = cmul(a, x); x.r += bu.x; x.i += bu.y;
        bp[(long)t * P_] = make_float2(x.r, x.i);
    }
}

// ---------- output heads: one wave per row ----------
__device__ __forceinline__ float dotcol(const float uv[16], int lane,
                                        const float* __restrict__ W, int ncol, int j) {
    float s = 0.0f;
#pragma unroll
    for (int i = 0; i < 16; ++i) s += uv[i] * W[(lane + 32 * i) * ncol + j];
#pragma unroll
    for (int off = 16; off > 0; off >>= 1) s += __shfl_xor(s, off, 32);
    return s;
}

__global__ __launch_bounds__(256)
void heads(const float* __restrict__ U,
           const float* __restrict__ dW, const float* __restrict__ db,
           const float* __restrict__ hW, const float* __restrict__ hb,
           const float* __restrict__ sW, const float* __restrict__ sb,
           const float* __restrict__ pW, const float* __restrict__ pb,
           const float* __restrict__ cW, const float* __restrict__ cb,
           float* __restrict__ out) {
    int lane = threadIdx.x & 31;
    long row = (long)blockIdx.x * 8 + (threadIdx.x >> 5);
    const float* ur = U + row * (long)H_;
    float uv[16];
#pragma unroll
    for (int i = 0; i < 16; ++i) uv[i] = ur[lane + 32 * i];

    const long OFF_POS    = 0;
    const long OFF_COMBO  = (long)M_ * 2;
    const long OFF_CURVE  = OFF_COMBO + M_;
    const long OFF_TIMING = OFF_CURVE + M_;
    const long OFF_REP    = OFF_TIMING + M_;
    const long OFF_HIT    = OFF_REP + M_;
    const long OFF_SLIDER = OFF_HIT + (long)M_ * 4;
    const long OFF_DIFF   = OFF_SLIDER + (long)M_ * 4;

    for (int j = 0; j < 2; ++j) {
        float s = dotcol(uv, lane, pW, 2, j) + pb[j];
        if (lane == 0) out[OFF_POS + row * 2 + j] = tanhf(s);
    }
    for (int j = 0; j < 4; ++j) {
        float s = dotcol(uv, lane, hW, 4, j) + hb[j];
        if (lane == 0) out[OFF_HIT + row * 4 + j] = s;
    }
    for (int j = 0; j < 4; ++j) {
        float s = dotcol(uv, lane, sW, 4, j) + sb[j];
        if (lane == 0) out[OFF_SLIDER + row * 4 + j] = s;
    }
    for (int j = 0; j < 4; ++j) {
        float s = dotcol(uv, lane, dW, 4, j) + db[j];
        if (lane == 0) out[OFF_DIFF + row * 4 + j] = tanhf(s);
    }
    {
        float s0 = dotcol(uv, lane, cW, 4, 0) + cb[0];
        float s1 = dotcol(uv, lane, cW, 4, 1) + cb[1];
        float s2 = dotcol(uv, lane, cW, 4, 2) + cb[2];
        float s3 = dotcol(uv, lane, cW, 4, 3) + cb[3];
        if (lane == 0) {
            out[OFF_TIMING + row] = s0;
            out[OFF_COMBO + row]  = s1;
            out[OFF_CURVE + row]  = s2;
            out[OFF_REP + row]    = fmaxf(exp2f(fmaxf(s3, 0.0f)), 1.0f);
        }
    }
}

// ---------- host orchestration ----------
extern "C" void kernel_launch(void* const* d_in, const int* in_sizes, int n_in,
                              void* d_out, int out_size, void* d_ws, size_t ws_size,
                              hipStream_t stream) {
    (void)in_sizes; (void)n_in; (void)out_size; (void)ws_size;
    const float* raw     = (const float*)d_in[0];
    const int*   seq     = (const int*)d_in[1];
    const float* diffr   = (const float*)d_in[2];
    const float* favr    = (const float*)d_in[3];
    const float* in_W    = (const float*)d_in[4];
    const float* in_b    = (const float*)d_in[5];
    const float* shift_W = (const float*)d_in[6];
    const float* shift_b = (const float*)d_in[7];
    const float* scale_W = (const float*)d_in[8];
    const float* scale_b = (const float*)d_in[9];
    const float* lam_re  = (const float*)d_in[10];
    const float* lam_im  = (const float*)d_in[11];
    const float* log_dt  = (const float*)d_in[12];
    const float* B_re    = (const float*)d_in[13];
    const float* B_im    = (const float*)d_in[14];
    const float* C_re    = (const float*)d_in[15];
    const float* C_im    = (const float*)d_in[16];
    const float* Dp      = (const float*)d_in[17];
    const float* W1      = (const float*)d_in[18];
    const float* b1      = (const float*)d_in[19];
    const float* W2      = (const float*)d_in[20];
    const float* b2      = (const float*)d_in[21];
    const float* dW      = (const float*)d_in[22];
    const float* db_     = (const float*)d_in[23];
    const float* hW      = (const float*)d_in[24];
    const float* hb      = (const float*)d_in[25];
    const float* sW      = (const float*)d_in[26];
    const float* sb      = (const float*)d_in[27];
    const float* pW      = (const float*)d_in[28];
    const float* pb      = (const float*)d_in[29];
    const float* cW      = (const float*)d_in[30];
    const float* cb      = (const float*)d_in[31];

    char* w = (char*)d_ws;
    size_t off = 0;
    auto alloc = [&](size_t bytes) -> void* {
        void* p = w + off;
        off = (off + bytes + 255) & ~(size_t)255;
        return p;
    };
    float*          uA     = (float*)alloc((size_t)M_ * H_ * 4);           // 64 MB
    float*          zB     = (float*)alloc((size_t)M_ * H_ * 4);           // 64 MB
    float*          xC     = (float*)alloc((size_t)M_ * 2 * P_ * 4);       // 64 MB (Bu -> x)
    float*          ffD    = (float*)alloc((size_t)M_ * 2 * H_ * 4);       // 128 MB (also xn)
    int*            rst    = (int*)alloc((size_t)B_ * L_ * 4);
    float*          lamBar = (float*)alloc((size_t)DEPTH_ * P_ * 2 * 4);
    unsigned short* Wbut   = (unsigned short*)alloc((size_t)DEPTH_ * 2 * P_ * H_ * 2);
    unsigned short* Cwt    = (unsigned short*)alloc((size_t)DEPTH_ * H_ * 2 * P_ * 2);
    unsigned short* inWt   = (unsigned short*)alloc((size_t)H_ * F_ * 2);
    unsigned short* W1t    = (unsigned short*)alloc((size_t)DEPTH_ * 2 * H_ * H_ * 2);
    unsigned short* W2t    = (unsigned short*)alloc((size_t)DEPTH_ * H_ * 2 * H_ * 2);
    float*          carryA = (float*)alloc((size_t)B_ * NCH * P_ * 2 * 4);
    float*          carryX = (float*)alloc((size_t)B_ * NCH * P_ * 2 * 4);
    float*          shiftA = (float*)alloc((size_t)DEPTH_ * B_ * H_ * 4);
    float*          scaleA = (float*)alloc((size_t)DEPTH_ * B_ * H_ * 4);
    float*          xn     = ffD;  // alias: xn only needed before layer loop

    // one-time prep: all GEMM B-operands stored transposed [N][K] in bf16
    { long n = (long)F_ * H_;
      cvt_bf16_t_batched<<<(n + 255) / 256, 256, 0, stream>>>(in_W, inWt, F_, H_, 1); }
    { long n = (long)DEPTH_ * H_ * 2 * H_;
      cvt_bf16_t_batched<<<(n + 255) / 256, 256, 0, stream>>>(W1, W1t, H_, 2 * H_, DEPTH_); }
    { long n = (long)DEPTH_ * 2 * H_ * H_;
      cvt_bf16_t_batched<<<(n + 255) / 256, 256, 0, stream>>>(W2, W2t, 2 * H_, H_, DEPTH_); }
    { long n = (long)DEPTH_ * P_ * H_;
      build_weights<<<(n + 255) / 256, 256, 0, stream>>>(lam_re, lam_im, log_dt, B_re, B_im,
                                                         C_re, C_im, Wbut, Cwt, lamBar); }
    { long n = (long)DEPTH_ * B_ * H_;
      modfilm<<<(n + 255) / 256, 256, 0, stream>>>(diffr, favr, shift_W, shift_b,
                                                   scale_W, scale_b, shiftA, scaleA); }
    { long n = (long)B_ * L_; mk_reset<<<(n + 255) / 256, 256, 0, stream>>>(seq, rst); }

    // xn = std(raw_audio); u = xn @ in_W + in_b
    rowln<false><<<M_, 256, 0, stream>>>(raw, xn, nullptr, nullptr);
    gemm_wmma<0, true><<<dim3(H_ / 128, M_ / 64), 256, 0, stream>>>(xn, inWt, in_b, nullptr, nullptr,
                                                                    uA, M_, H_, F_);

    for (int l = 0; l < DEPTH_; ++l) {
        // Bu (re/im interleaved) = u @ Wbut[l]^T    (N=2P, K=H)
        gemm_wmma<0, false><<<dim3(2 * P_ / 128, M_ / 64), 256, 0, stream>>>(
            uA, Wbut + (size_t)l * 2 * P_ * H_, nullptr, nullptr, nullptr, xC, M_, 2 * P_, H_);
        // chunked complex scan in-place on xC
        scan_phase1<<<B_ * NCH, P_, 0, stream>>>(xC, rst, lamBar + (size_t)l * P_ * 2, carryA, carryX);
        scan_phase2<<<(B_ * P_ + 255) / 256, 256, 0, stream>>>(carryA, carryX);
        scan_phase3<<<B_ * NCH, P_, 0, stream>>>(xC, rst, lamBar + (size_t)l * P_ * 2, carryA);
        // z = u + gelu(Re(x C) + D*u)               (N=H, K=2P)
        gemm_wmma<2, false><<<dim3(H_ / 128, M_ / 64), 256, 0, stream>>>(
            xC, Cwt + (size_t)l * H_ * 2 * P_, nullptr, uA, Dp + (size_t)l * H_, zB, M_, H_, 2 * P_);
        // ffh = gelu(z @ W1 + b1)                   (N=2H, K=H)
        gemm_wmma<1, true><<<dim3(2 * H_ / 128, M_ / 64), 256, 0, stream>>>(
            zB, W1t + (size_t)l * 2 * H_ * H_, b1 + (size_t)l * 2 * H_, nullptr, nullptr,
            ffD, M_, 2 * H_, H_);
        // t = z + ffh @ W2 + b2  (into uA), then u = std(t)*scale + shift (in place)
        gemm_wmma<3, true><<<dim3(H_ / 128, M_ / 64), 256, 0, stream>>>(
            ffD, W2t + (size_t)l * H_ * 2 * H_, b2 + (size_t)l * H_, zB, nullptr, uA, M_, H_, 2 * H_);
        rowln<true><<<M_, 256, 0, stream>>>(uA, uA, scaleA + (size_t)l * B_ * H_,
                                            shiftA + (size_t)l * B_ * H_);
    }

    heads<<<M_ / 8, 256, 0, stream>>>(uA, dW, db_, hW, hb, sW, sb, pW, pb, cW, cb, (float*)d_out);
}
